// MultiShapeKernel_75453985457456
// MI455X (gfx1250) — matrix operand order repaired and checked by reading, not compiled, not run on hardware
//
#include <hip/hip_runtime.h>
#include <hip/hip_bf16.h>
#include <math.h>

// Problem constants (from reference setup_inputs)
#define BN_N    8
#define BN_C    128
#define BN_H    128
#define BN_W    128
#define PLANE   (BN_H * BN_W)        // 16384
#define NPLANES (BN_N * BN_C)        // 1024
#define EPS     1e-5f

typedef __attribute__((ext_vector_type(16))) _Float16 v16h;
typedef __attribute__((ext_vector_type(8)))  float    v8f;
typedef __attribute__((ext_vector_type(4)))  float    v4f;   // for NT stores

__device__ __forceinline__ int refl(int i, int n) {
    // jnp.pad mode='reflect' (no edge repeat): -1 -> 1, n -> n-2
    return (i < 0) ? -i : ((i >= n) ? (2 * n - 2 - i) : i);
}

// Per-row halo exchange inside one wave32: lane owns 4 consecutive w values
// (float4), full 128-wide row per wave. Left/right neighbors come from
// adjacent lanes via shuffle; row ends use reflect padding.
__device__ __forceinline__ void row_halo(const float4& v, int lane,
                                         float& left, float& right) {
    left  = __shfl_up(v.w, 1, 32);
    right = __shfl_down(v.x, 1, 32);
    if (lane == 0)  left  = v.y;   // w=-1  reflects to w=1
    if (lane == 31) right = v.z;   // w=128 reflects to w=126
}

// 3-tap conv along w applied to a lane's 4 elements, halos supplied.
__device__ __forceinline__ float4 row_conv(const float4& v, float l, float r,
                                           float f0, float f1, float f2) {
    float4 o;
    o.x = f0 * l   + f1 * v.x + f2 * v.y;
    o.y = f0 * v.x + f1 * v.y + f2 * v.z;
    o.z = f0 * v.y + f1 * v.z + f2 * v.w;
    o.w = f0 * v.z + f1 * v.w + f2 * r;
    return o;
}

// ---------------------------------------------------------------------------
// Kernel 1: per-(n,c) plane reductions over x.
//   mean_x[plane] = mean over 128x128; gap_h[plane][h] = row mean over w.
// One block (256 thr = 8 wave32) per plane; a wave reads one row per step as
// 32 x b128 (512B/wave), shuffle-tree reduces it.
// ---------------------------------------------------------------------------
__global__ void reduce_plane_kernel(const float* __restrict__ x,
                                    float* __restrict__ mean_x,
                                    float* __restrict__ gap_h) {
    const int plane = blockIdx.x;
    const float4* p4 = (const float4*)(x + (size_t)plane * PLANE);
    const int tid  = threadIdx.x;
    const int lane = tid & 31;
    const int wave = tid >> 5;
    __shared__ float s_wave[8];

    float wave_acc = 0.f;
    for (int r = 0; r < 16; ++r) {
        const int h = wave + (r << 3);
        const float4 v = p4[h * 32 + lane];
        float s = (v.x + v.y) + (v.z + v.w);
        #pragma unroll
        for (int off = 16; off > 0; off >>= 1) s += __shfl_xor(s, off, 32);
        if (lane == 0) {
            gap_h[(size_t)plane * BN_H + h] = s * (1.f / 128.f);
            wave_acc += s;
        }
    }
    if (lane == 0) s_wave[wave] = wave_acc;
    __syncthreads();
    if (tid == 0) {
        float t = 0.f;
        #pragma unroll
        for (int i = 0; i < 8; ++i) t += s_wave[i];
        mean_x[plane] = t * (1.f / (float)PLANE);
    }
}

// ---------------------------------------------------------------------------
// Kernel 2: dynamic-filter GEMM via WMMA.
//   out[m][o] = tanh( maybeBN( sum_k A[m][k] * Wt[o][k] ) ),  m<8, k<128, o<OUT
// One wave32 per block, one 16-wide N-tile per block, K=128 as 4 chained
// v_wmma_f32_16x16x32_f16. Fragment layouts used here (A/B in contiguous k order, same for both):
//   A 16x32 f16 : lane -> M = lane&15 ; elem e -> K = kb + (lane<16?0:16)
//                 + e
//   B 32x16 f16 : lane -> N = lane&15 ; elem e -> K = kb + (lane<16?0:16) + e
//   D 16x16 f32 : elem r -> M = r + (lane<16?0:8) ; N = lane&15
// Masking uses branchless selects on clamped addresses (EXEC stays full).
// ---------------------------------------------------------------------------
__global__ void filt_gemm_kernel(const float* __restrict__ A,      // (8,128)
                                 const float* __restrict__ Wt,     // (OUT,128)
                                 float* __restrict__ out,          // (8,OUT)
                                 int OUT, int use_bn,
                                 const float* __restrict__ bn_scale,
                                 const float* __restrict__ bn_bias,
                                 const float* __restrict__ bn_mean,
                                 const float* __restrict__ bn_var) {
    const int lane = threadIdx.x;           // 0..31, one wave
    const int tile = blockIdx.x;            // N-tile of 16 columns
    const int mrow = lane & 15;             // A row (M)
    const int ncol = tile * 16 + (lane & 15);

    const int   arow  = mrow & 7;                       // clamped (valid) row
    const float amask = (mrow < 8) ? 1.f : 0.f;
    const int   bcol  = (ncol < OUT) ? ncol : 0;        // clamped column
    const float bmask = (ncol < OUT) ? 1.f : 0.f;

    v8f acc = {};
    for (int kt = 0; kt < 4; ++kt) {
        const int kbase = kt * 32;
        v16h a, b;
        const int aklo = kbase + ((lane < 16) ? 0 : 16);
        #pragma unroll
        for (int e = 0; e < 16; ++e) {
            const int k = aklo + e;
            a[e] = (_Float16)(A[arow * 128 + k] * amask);
        }
        const int bklo = kbase + ((lane < 16) ? 0 : 16);
        #pragma unroll
        for (int e = 0; e < 16; ++e) {
            b[e] = (_Float16)(Wt[bcol * 128 + bklo + e] * bmask);
        }
        acc = __builtin_amdgcn_wmma_f32_16x16x32_f16(
                  false, a, false, b, (short)0, acc, false, false);
    }

    // Scatter D. Only lanes 0..15 hold rows M=0..7 (lanes>=16 hold M=r+8>=8,
    // never real) -> single uniform-per-lane guard, no per-r exec churn.
    if (lane < 16 && ncol < OUT) {
        float bnA = 1.f, bnB = 0.f;
        if (use_bn) {
            bnA = rsqrtf(bn_var[ncol] + EPS) * bn_scale[ncol];
            bnB = bn_bias[ncol] - bn_mean[ncol] * bnA;
        }
        #pragma unroll
        for (int r = 0; r < 8; ++r) {
            out[r * OUT + ncol] = tanhf(acc[r] * bnA + bnB);
        }
    }
}

// ---------------------------------------------------------------------------
// Kernel 3: horizontal strip attention, fused with reductions of its output.
//   out_h = (conv_w(x)*(ins+1) - ins*gap_h)*lamb_l + x*(lamb_h+1)
// Wave owns a full 128-wide row as 32 x float4 (b128 traffic); horizontal
// halos via intra-wave shuffle. Also produces gap_w (column means of out_h)
// and mean_outh per plane.
// ---------------------------------------------------------------------------
__global__ void strip_h_kernel(const float* __restrict__ x,
                               const float* __restrict__ filt_h,   // (8,24)
                               const float* __restrict__ gap_h,    // (1024,128)
                               const float* __restrict__ h_inside,
                               const float* __restrict__ h_lamb_l,
                               const float* __restrict__ h_lamb_h,
                               float* __restrict__ out_h,
                               float* __restrict__ gap_w,          // (1024,128)
                               float* __restrict__ mean_outh) {    // (1024)
    const int plane = blockIdx.x;
    const int n = plane >> 7, c = plane & 127, g = c >> 4;
    const float f0 = filt_h[n * 24 + g * 3 + 0];
    const float f1 = filt_h[n * 24 + g * 3 + 1];
    const float f2 = filt_h[n * 24 + g * 3 + 2];
    const float ins = h_inside[c];
    const float ll  = h_lamb_l[c];
    const float lh1 = h_lamb_h[c] + 1.f;
    const float ip1 = ins + 1.f;

    const float4* px4 = (const float4*)(x     + (size_t)plane * PLANE);
    float4*       po4 = (float4*)      (out_h + (size_t)plane * PLANE);
    const float*  pg  = gap_h + (size_t)plane * BN_H;

    const int tid  = threadIdx.x;
    const int lane = tid & 31;
    const int wave = tid >> 5;

    __shared__ float4 s4[8][32];     // per-wave column partial sums
    __shared__ float  sred[128];

    float4 colsum = make_float4(0.f, 0.f, 0.f, 0.f);
    for (int r = 0; r < 16; ++r) {
        const int h = wave + (r << 3);
        const float4 v = px4[h * 32 + lane];
        float left, right;
        row_halo(v, lane, left, right);
        const float4 conv = row_conv(v, left, right, f0, f1, f2);
        const float  gh   = pg[h] * ins;
        float4 val;
        val.x = (conv.x * ip1 - gh) * ll + v.x * lh1;
        val.y = (conv.y * ip1 - gh) * ll + v.y * lh1;
        val.z = (conv.z * ip1 - gh) * ll + v.z * lh1;
        val.w = (conv.w * ip1 - gh) * ll + v.w * lh1;
        po4[h * 32 + lane] = val;    // RT store: out_h stays L2-resident
        colsum.x += val.x; colsum.y += val.y;
        colsum.z += val.z; colsum.w += val.w;
    }
    s4[wave][lane] = colsum;
    __syncthreads();

    if (tid < 128) {                 // w = tid
        const float* sf = (const float*)s4;
        float cs = 0.f;
        #pragma unroll
        for (int wv = 0; wv < 8; ++wv) cs += sf[wv * 128 + tid];
        gap_w[(size_t)plane * BN_W + tid] = cs * (1.f / 128.f);
        sred[tid] = cs;
    }
    __syncthreads();
    for (int off = 64; off > 0; off >>= 1) {
        if (tid < off) sred[tid] += sred[tid + off];
        __syncthreads();
    }
    if (tid == 0) mean_outh[plane] = sred[0] * (1.f / (float)PLANE);
}

// ---------------------------------------------------------------------------
// Kernel 4: vertical strip on out_h + 3x3 square attention on x + combine.
//   out = gamma*out_hw + beta*x + square
// 6 b128 loads per 4 outputs (3 rows of out_h + 3 rows of x); horizontal
// halos via shuffle. Output stored nontemporal (never re-read) so x/out_h
// stay resident in the 192MB L2.
// ---------------------------------------------------------------------------
__global__ void final_kernel(const float* __restrict__ x,
                             const float* __restrict__ out_h,
                             const float* __restrict__ filt_w,     // (8,24)
                             const float* __restrict__ gap_w,      // (1024,128)
                             const float* __restrict__ w_inside,
                             const float* __restrict__ w_lamb_l,
                             const float* __restrict__ w_lamb_h,
                             const float* __restrict__ filt_sq,    // (8,72)
                             const float* __restrict__ mean_x,     // (1024)
                             const float* __restrict__ sq_inside,
                             const float* __restrict__ sq_lamb_l,
                             const float* __restrict__ sq_lamb_h,
                             const float* __restrict__ gamma,
                             const float* __restrict__ beta,
                             float* __restrict__ out) {
    const int plane = blockIdx.x;
    const int n = plane >> 7, c = plane & 127, g = c >> 4;

    const float fw0 = filt_w[n * 24 + g * 3 + 0];
    const float fw1 = filt_w[n * 24 + g * 3 + 1];
    const float fw2 = filt_w[n * 24 + g * 3 + 2];
    float fs[9];
    #pragma unroll
    for (int k = 0; k < 9; ++k) fs[k] = filt_sq[n * 72 + g * 9 + k];

    const float insW = w_inside[c],  llW = w_lamb_l[c],  lhW1 = w_lamb_h[c] + 1.f;
    const float insS = sq_inside[c], llS = sq_lamb_l[c], lhS1 = sq_lamb_h[c] + 1.f;
    const float gm = gamma[c], bt = beta[c];
    const float mxS = mean_x[plane] * insS;
    const float ipW = insW + 1.f, ipS = insS + 1.f;

    const float4* px4 = (const float4*)(x     + (size_t)plane * PLANE);
    const float4* ph4 = (const float4*)(out_h + (size_t)plane * PLANE);
    const float4* pg4 = (const float4*)(gap_w + (size_t)plane * BN_W);
    float4*       po4 = (float4*)      (out   + (size_t)plane * PLANE);

    const int tid  = threadIdx.x;
    const int lane = tid & 31;
    const int wave = tid >> 5;

    const float4 gw = pg4[lane];     // lane-constant across rows
    const float4 gwi = make_float4(gw.x * insW, gw.y * insW,
                                   gw.z * insW, gw.w * insW);

    for (int r = 0; r < 16; ++r) {
        const int h  = wave + (r << 3);
        const int hm = refl(h - 1, BN_H);
        const int hp = refl(h + 1, BN_H);

        // ---- vertical strip on out_h (no horizontal halo needed) ----
        const float4 oc = ph4[h  * 32 + lane];
        const float4 om = ph4[hm * 32 + lane];
        const float4 op = ph4[hp * 32 + lane];
        float4 ohw;
        ohw.x = ((fw0*om.x + fw1*oc.x + fw2*op.x) * ipW - gwi.x) * llW + oc.x * lhW1;
        ohw.y = ((fw0*om.y + fw1*oc.y + fw2*op.y) * ipW - gwi.y) * llW + oc.y * lhW1;
        ohw.z = ((fw0*om.z + fw1*oc.z + fw2*op.z) * ipW - gwi.z) * llW + oc.z * lhW1;
        ohw.w = ((fw0*om.w + fw1*oc.w + fw2*op.w) * ipW - gwi.w) * llW + oc.w * lhW1;

        // ---- 3x3 square attention on x ----
        const float4 xc = px4[h  * 32 + lane];
        const float4 xm = px4[hm * 32 + lane];
        const float4 xp = px4[hp * 32 + lane];
        float lm, rm, lc, rc, lp, rp;
        row_halo(xm, lane, lm, rm);
        row_halo(xc, lane, lc, rc);
        row_halo(xp, lane, lp, rp);
        const float4 sA = row_conv(xm, lm, rm, fs[0], fs[1], fs[2]);
        const float4 sB = row_conv(xc, lc, rc, fs[3], fs[4], fs[5]);
        const float4 sC = row_conv(xp, lp, rp, fs[6], fs[7], fs[8]);

        v4f o;
        o.x = gm*ohw.x + xc.x*bt + ((sA.x+sB.x+sC.x) * ipS - mxS) * llS + xc.x*lhS1;
        o.y = gm*ohw.y + xc.y*bt + ((sA.y+sB.y+sC.y) * ipS - mxS) * llS + xc.y*lhS1;
        o.z = gm*ohw.z + xc.z*bt + ((sA.z+sB.z+sC.z) * ipS - mxS) * llS + xc.z*lhS1;
        o.w = gm*ohw.w + xc.w*bt + ((sA.w+sB.w+sC.w) * ipS - mxS) * llS + xc.w*lhS1;

        // NT store through a clang ext_vector pointer (builtin requires
        // scalar-or-ext-vector, not HIP's float4 class type).
        __builtin_nontemporal_store(o, (v4f*)(po4 + h * 32 + lane));
    }
}

// ---------------------------------------------------------------------------
extern "C" void kernel_launch(void* const* d_in, const int* in_sizes, int n_in,
                              void* d_out, int out_size, void* d_ws, size_t ws_size,
                              hipStream_t stream) {
    const float* x           = (const float*)d_in[0];
    const float* sq_conv_w   = (const float*)d_in[1];   // (72,128)
    const float* sq_bn_scale = (const float*)d_in[2];
    const float* sq_bn_bias  = (const float*)d_in[3];
    const float* sq_bn_mean  = (const float*)d_in[4];
    const float* sq_bn_var   = (const float*)d_in[5];
    const float* sq_inside   = (const float*)d_in[6];
    const float* sq_lamb_l   = (const float*)d_in[7];
    const float* sq_lamb_h   = (const float*)d_in[8];
    const float* h_conv_w    = (const float*)d_in[9];   // (24,128)
    const float* h_inside    = (const float*)d_in[10];
    const float* h_lamb_l    = (const float*)d_in[11];
    const float* h_lamb_h    = (const float*)d_in[12];
    const float* w_conv_w    = (const float*)d_in[13];  // (24,128)
    const float* w_inside    = (const float*)d_in[14];
    const float* w_lamb_l    = (const float*)d_in[15];
    const float* w_lamb_h    = (const float*)d_in[16];
    const float* gamma       = (const float*)d_in[17];
    const float* beta        = (const float*)d_in[18];
    float* out = (float*)d_out;

    // Workspace layout (floats); every offset is a multiple of 4 floats
    // so float4 views stay 16B-aligned.
    float* ws = (float*)d_ws;
    float* ws_out_h   = ws;                                   // 16,777,216
    float* ws_mean_x  = ws_out_h  + (size_t)NPLANES * PLANE;  // 1024
    float* ws_gap_h   = ws_mean_x + NPLANES;                  // 131072
    float* ws_gap_w   = ws_gap_h  + NPLANES * BN_H;           // 131072
    float* ws_mean_oh = ws_gap_w  + NPLANES * BN_W;           // 1024
    float* ws_filt_h  = ws_mean_oh + NPLANES;                 // 8*24
    float* ws_filt_w  = ws_filt_h  + BN_N * 24;               // 8*24
    float* ws_filt_sq = ws_filt_w  + BN_N * 24;               // 8*72

    // 1) reductions over x
    reduce_plane_kernel<<<NPLANES, 256, 0, stream>>>(x, ws_mean_x, ws_gap_h);

    // 2) dynamic filters from GAP(x): WMMA GEMMs (N tiles of 16)
    filt_gemm_kernel<<<2, 32, 0, stream>>>(ws_mean_x, h_conv_w, ws_filt_h,
                                           24, 0, nullptr, nullptr, nullptr, nullptr);
    filt_gemm_kernel<<<5, 32, 0, stream>>>(ws_mean_x, sq_conv_w, ws_filt_sq,
                                           72, 1, sq_bn_scale, sq_bn_bias,
                                           sq_bn_mean, sq_bn_var);

    // 3) horizontal strip pass (+ out_h reductions)
    strip_h_kernel<<<NPLANES, 256, 0, stream>>>(x, ws_filt_h, ws_gap_h,
                                                h_inside, h_lamb_l, h_lamb_h,
                                                ws_out_h, ws_gap_w, ws_mean_oh);

    // 4) vertical-strip filter from GAP(out_h): WMMA GEMM
    filt_gemm_kernel<<<2, 32, 0, stream>>>(ws_mean_oh, w_conv_w, ws_filt_w,
                                           24, 0, nullptr, nullptr, nullptr, nullptr);

    // 5) vertical strip + square attention + combine
    final_kernel<<<NPLANES, 256, 0, stream>>>(x, ws_out_h,
                                              ws_filt_w, ws_gap_w,
                                              w_inside, w_lamb_l, w_lamb_h,
                                              ws_filt_sq, ws_mean_x,
                                              sq_inside, sq_lamb_l, sq_lamb_h,
                                              gamma, beta, out);
}
